// ClusterGuidedMambaBlock_50826642981202
// MI455X (gfx1250) — compile-verified, matrix operand order checked by reading, not run on hardware
//
#include <hip/hip_runtime.h>
#include <hip/hip_bf16.h>

// ---------------------------------------------------------------------------
// ClusterGuidedMambaBlock on gfx1250 (MI455X).
// GEMMs: bf16 WMMA (f32 accumulate), templated shapes, double-buffered LDS
// tiles with async-to-LDS pipeline (ASYNCcnt) when the toolchain exposes it.
// ---------------------------------------------------------------------------

#define D_MODEL   768
#define D_STATE   16
#define D_CONV    4
#define D_INNER   1536
#define TWO_INNER 3072
#define D_HID     384
#define BATCH     2
#define SEQ       2048
#define NROWS     (BATCH * SEQ)   // 4096

typedef __attribute__((ext_vector_type(16))) __bf16 v16bf;
typedef __attribute__((ext_vector_type(8)))  float  v8f;
typedef __attribute__((ext_vector_type(4)))  int    v4i_t;

typedef __attribute__((address_space(1))) v4i_t* gv4i_p;   // global int4*
typedef __attribute__((address_space(3))) v4i_t* lv4i_p;   // LDS int4*

#if defined(__has_builtin)
#if __has_builtin(__builtin_amdgcn_global_load_async_to_lds_b128)
#define HAVE_ASYNC_LDS 1
#endif
#endif
#ifndef HAVE_ASYNC_LDS
#define HAVE_ASYNC_LDS 0
#endif

__device__ __forceinline__ void wait_asynccnt0() {
#if defined(__has_builtin)
#if __has_builtin(__builtin_amdgcn_s_wait_asynccnt)
    __builtin_amdgcn_s_wait_asynccnt(0);
    return;
#else
    asm volatile("s_wait_asynccnt 0x0" ::: "memory");
    return;
#endif
#else
    asm volatile("s_wait_asynccnt 0x0" ::: "memory");
#endif
}

// ---- small device helpers -------------------------------------------------

__device__ __forceinline__ unsigned short f2bf(float f) {
    unsigned int u = __float_as_uint(f);
    unsigned int r = (u + 0x7FFFu + ((u >> 16) & 1u)) >> 16;
    return (unsigned short)r;
}

__device__ __forceinline__ float sigmoidf_(float x) {
    return 1.0f / (1.0f + __expf(-x));
}

__device__ __forceinline__ float softplusf_(float x) {
    return (x > 20.0f) ? x : log1pf(__expf(x));
}

// ---- f32 -> bf16 conversion ------------------------------------------------

__global__ void cvt_bf16_kernel(const float* __restrict__ in,
                                unsigned short* __restrict__ out, int n) {
    int i = blockIdx.x * 256 + threadIdx.x;
    if (i < n) out[i] = f2bf(in[i]);
}

// ---- LayerNorm -> bf16 -----------------------------------------------------

__global__ __launch_bounds__(256)
void layernorm_kernel(const float* __restrict__ x,
                      const float* __restrict__ g,
                      const float* __restrict__ b,
                      unsigned short* __restrict__ out) {
    int r = blockIdx.x;
    int tid = threadIdx.x;
    const float* row = x + (size_t)r * D_MODEL;
    float lsum = 0.f, lsq = 0.f;
    for (int i = tid; i < D_MODEL; i += 256) {
        float v = row[i];
        lsum += v;
        lsq  += v * v;
    }
    __shared__ float s1[256], s2[256];
    s1[tid] = lsum; s2[tid] = lsq;
    __syncthreads();
    for (int off = 128; off > 0; off >>= 1) {
        if (tid < off) { s1[tid] += s1[tid + off]; s2[tid] += s2[tid + off]; }
        __syncthreads();
    }
    float mu   = s1[0] * (1.0f / D_MODEL);
    float var  = s2[0] * (1.0f / D_MODEL) - mu * mu;
    float rinv = rsqrtf(var + 1e-5f);
    for (int i = tid; i < D_MODEL; i += 256) {
        float v = (row[i] - mu) * rinv * g[i] + b[i];
        out[(size_t)r * D_MODEL + i] = f2bf(v);
    }
}

// ---- generic bf16 WMMA GEMM ------------------------------------------------
// C[M,N] = A[M,K] (bf16, row-major) x B[K,N] (bf16, row-major), f32 accumulate.
// Block tile 128x128, 8 waves as 4(M) x 2(N); wave tile 32x64 = 2x4 WMMA tiles
// (8 v_wmma per K-step of 32). Double-buffered LDS stages; A tile streamed via
// global_load_async_to_lds_b128 when available. N, K, MODE compile-time.
// MODE 0: C = AB
// MODE 1: C = sigmoid(AB + aux[n])            (aux = bias over N)
// MODE 2: C = AB + aux[m*N + n]               (aux = residual, full matrix)

#define LDS_PITCH 40   // halves; 80B row pitch -> conflict-free, 16B-aligned chunks

__device__ __forceinline__ v16bf load_frag(const unsigned short* lds,
                                           int rowIdx, int lane) {
    int kb   = (lane & 16) ? 8 : 0;
    int base = rowIdx * LDS_PITCH + kb;   // byte addr = rowIdx*80 + {0,16}: 16B aligned
    union { v16bf v; uint4 q[2]; } u;
    u.q[0] = *(const uint4*)(lds + base);        // K halves kb .. kb+7
    u.q[1] = *(const uint4*)(lds + base + 16);   // K halves kb+16 .. kb+23
    return u.v;
}

template <int N, int K, int MODE>
__global__ __launch_bounds__(256)
void wmma_gemm_kernel(const unsigned short* __restrict__ A,
                      const unsigned short* __restrict__ B,
                      float* __restrict__ C,
                      const float* __restrict__ aux) {
    __shared__ unsigned short lA[2][128 * LDS_PITCH];
    __shared__ unsigned short lB[2][128 * LDS_PITCH];

    const int tid  = threadIdx.x;
    const int lane = tid & 31;
    const int wave = tid >> 5;
    const int wm   = wave >> 1;      // 0..3  -> 32-row band
    const int wn   = wave & 1;       // 0..1  -> 64-col band
    const int rowBase = blockIdx.y * 128;
    const int colBase = blockIdx.x * 128;
    constexpr int NK = K / 32;

    v8f acc[2][4];
    #pragma unroll
    for (int i = 0; i < 2; ++i)
        #pragma unroll
        for (int j = 0; j < 4; ++j)
            #pragma unroll
            for (int e = 0; e < 8; ++e)
                acc[i][j][e] = 0.0f;

    // A loader: 128 rows x 32 halves; thread -> (row, 16-half chunk)
    const int lr  = tid >> 1;
    const int lch = tid & 1;
    // B loader: transpose 32(K) x 128(N) into lB[n][k]
    const int bn = tid & 127;
    const int bk = (tid >> 7) * 16;

    const unsigned short* gArow = A + (size_t)(rowBase + lr) * K + lch * 16;
    const int dAoff = lr * LDS_PITCH + lch * 16;

    // Stage issue: fill LDS buffer `buf` with K-slice starting at k0.
    auto issue_load = [&](int buf, int k0) {
        if (k0 + 32 < K) {
            __builtin_prefetch(gArow + k0 + 32, 0, 3);
            __builtin_prefetch(B + (size_t)(k0 + 32 + bk) * N + colBase + bn, 0, 3);
        }
#if HAVE_ASYNC_LDS
        {
            gv4i_p gsrc = (gv4i_p)(gArow + k0);
            lv4i_p ldst = (lv4i_p)(&lA[buf][dAoff]);
            __builtin_amdgcn_global_load_async_to_lds_b128(gsrc, ldst, 0, 0);
            __builtin_amdgcn_global_load_async_to_lds_b128(gsrc, ldst, 16, 0);
        }
#else
        {
            const uint4* gA = (const uint4*)(gArow + k0);
            uint4* dA = (uint4*)(&lA[buf][dAoff]);
            dA[0] = gA[0];
            dA[1] = gA[1];
        }
#endif
        #pragma unroll
        for (int i = 0; i < 16; ++i)
            lB[buf][bn * LDS_PITCH + bk + i] =
                B[(size_t)(k0 + bk + i) * N + colBase + bn];
    };

    issue_load(0, 0);

    for (int ik = 0; ik < NK; ++ik) {
        const int buf = ik & 1;
#if HAVE_ASYNC_LDS
        wait_asynccnt0();
#endif
        __syncthreads();   // stage `buf` fully resident; previous compute done
        if (ik + 1 < NK)
            issue_load(buf ^ 1, (ik + 1) * 32);

        v16bf af[2], bfr[4];
        #pragma unroll
        for (int tm = 0; tm < 2; ++tm)
            af[tm] = load_frag(lA[buf], wm * 32 + tm * 16 + (lane & 15), lane);
        #pragma unroll
        for (int tn = 0; tn < 4; ++tn)
            bfr[tn] = load_frag(lB[buf], wn * 64 + tn * 16 + (lane & 15), lane);

        #pragma unroll
        for (int tm = 0; tm < 2; ++tm)
            #pragma unroll
            for (int tn = 0; tn < 4; ++tn)
                acc[tm][tn] = __builtin_amdgcn_wmma_f32_16x16x32_bf16(
                    false, af[tm], false, bfr[tn], (short)0, acc[tm][tn],
                    false, false);
    }

    // Epilogue. C/D layout: VGPR v -> M = v + (lane>=16 ? 8:0), N = lane&15.
    const int mo = (lane & 16) ? 8 : 0;
    const int nl = lane & 15;
    #pragma unroll
    for (int tm = 0; tm < 2; ++tm) {
        #pragma unroll
        for (int tn = 0; tn < 4; ++tn) {
            const int m0 = rowBase + wm * 32 + tm * 16 + mo;
            const int n  = colBase + wn * 64 + tn * 16 + nl;
            float bias = (MODE == 1) ? aux[n] : 0.0f;
            float* cp = C + (size_t)m0 * N + n;
            const float* rp = (MODE == 2) ? (aux + (size_t)m0 * N + n) : nullptr;
            #pragma unroll
            for (int v = 0; v < 8; ++v) {
                float val = acc[tm][tn][v];
                if (MODE == 1)      val = sigmoidf_(val + bias);
                else if (MODE == 2) val += rp[(size_t)v * N];
                cp[(size_t)v * N] = val;
            }
        }
    }
}

// ---- depthwise causal conv (width 4) + bias + SiLU -------------------------

__global__ void conv_silu_kernel(const float* __restrict__ xz,
                                 const float* __restrict__ cw,
                                 const float* __restrict__ cb,
                                 float* __restrict__ xc) {
    int idx = blockIdx.x * 256 + threadIdx.x;
    if (idx >= NROWS * D_INNER) return;
    int c = idx % D_INNER;
    int r = idx / D_INNER;
    int t = r & (SEQ - 1);
    float s = cb[c];
    #pragma unroll
    for (int k = 0; k < D_CONV; ++k) {
        int tt = t - (D_CONV - 1) + k;
        if (tt >= 0)
            s += xz[(size_t)(r - t + tt) * TWO_INNER + c] * cw[c * D_CONV + k];
    }
    xc[idx] = s * sigmoidf_(s);   // SiLU
}

// ---- x_ssm = xc @ W_x  (1536 -> 33), row cached in LDS ---------------------

__global__ __launch_bounds__(64)
void xssm_kernel(const float* __restrict__ xc,
                 const float* __restrict__ Wx,
                 float* __restrict__ out) {
    __shared__ float row[D_INNER];
    int r = blockIdx.x;
    for (int i = threadIdx.x; i < D_INNER; i += 64)
        row[i] = xc[(size_t)r * D_INNER + i];
    __syncthreads();
    int j = threadIdx.x;
    if (j < 2 * D_STATE + 1) {
        float s = 0.f;
        for (int k = 0; k < D_INNER; ++k)
            s += row[k] * Wx[k * (2 * D_STATE + 1) + j];
        out[(size_t)r * (2 * D_STATE + 1) + j] = s;
    }
}

// ---- H = relu(importance * Wc1 + bc1) -> bf16 ------------------------------

__global__ void hproj_kernel(const float* __restrict__ imp,
                             const float* __restrict__ Wc1,
                             const float* __restrict__ bc1,
                             unsigned short* __restrict__ Hb) {
    int idx = blockIdx.x * 256 + threadIdx.x;
    if (idx >= NROWS * D_HID) return;
    int h = idx % D_HID;
    int r = idx / D_HID;
    float v = imp[r] * Wc1[h] + bc1[h];
    Hb[idx] = f2bf(v > 0.f ? v : 0.f);
}

// ---- delta = softplus(delta_raw * W_dt + b_dt) * (1 + mod), in place -------

__global__ void delta_kernel(const float* __restrict__ xssm,
                             const float* __restrict__ Wdt,
                             const float* __restrict__ bdt,
                             float* __restrict__ modbuf) {
    int idx = blockIdx.x * 256 + threadIdx.x;
    if (idx >= NROWS * D_INNER) return;
    int d = idx % D_INNER;
    int r = idx / D_INNER;
    float xr = xssm[(size_t)r * (2 * D_STATE + 1)];   // delta_raw
    float sp = softplusf_(xr * Wdt[d] + bdt[d]);
    modbuf[idx] = sp * (1.0f + modbuf[idx]);
}

// ---- selective scan: lane = (b, d, n); 16-lane shuffle reduce over n -------
// Produces yg = (y + xc*Dskip) * silu(z) directly in bf16.

__global__ __launch_bounds__(256)
void scan_kernel(const float* __restrict__ delta,
                 const float* __restrict__ xssm,
                 const float* __restrict__ xc,
                 const float* __restrict__ xz,       // z = cols [1536,3072)
                 const float* __restrict__ A_log,
                 const float* __restrict__ Dskip,
                 unsigned short* __restrict__ yg) {
    int gid = blockIdx.x * blockDim.x + threadIdx.x;
    int n = gid & 15;
    int g = gid >> 4;                 // 0 .. BATCH*D_INNER-1
    int d = g % D_INNER;
    int b = g / D_INNER;
    float Av = -__expf(A_log[d * D_STATE + n]);
    float Dv = Dskip[d];
    float h = 0.f;
    for (int t = 0; t < SEQ; ++t) {
        int r = b * SEQ + t;
        float dl = delta[(size_t)r * D_INNER + d];
        float Bt = xssm[(size_t)r * (2 * D_STATE + 1) + 1 + n];
        float Ct = xssm[(size_t)r * (2 * D_STATE + 1) + 1 + D_STATE + n];
        float xt = xc[(size_t)r * D_INNER + d];
        float dA = __expf(dl * Av);
        h = dA * h + (dl * xt) * Bt;
        float y = h * Ct;
        y += __shfl_xor(y, 1, 16);
        y += __shfl_xor(y, 2, 16);
        y += __shfl_xor(y, 4, 16);
        y += __shfl_xor(y, 8, 16);
        if (n == 0) {
            float yt = y + xt * Dv;
            float zv = xz[(size_t)r * TWO_INNER + D_INNER + d];
            float ys = yt * (zv * sigmoidf_(zv));
            yg[(size_t)r * D_INNER + d] = f2bf(ys);
        }
    }
}

// ---------------------------------------------------------------------------

static inline int cdiv(int a, int b) { return (a + b - 1) / b; }

extern "C" void kernel_launch(void* const* d_in, const int* in_sizes, int n_in,
                              void* d_out, int out_size, void* d_ws, size_t ws_size,
                              hipStream_t stream) {
    (void)in_sizes; (void)n_in; (void)out_size; (void)ws_size;

    const float* x       = (const float*)d_in[0];
    const float* imp     = (const float*)d_in[1];
    const float* ln_g    = (const float*)d_in[2];
    const float* ln_b    = (const float*)d_in[3];
    const float* W_in    = (const float*)d_in[4];
    const float* conv_w  = (const float*)d_in[5];
    const float* conv_b  = (const float*)d_in[6];
    const float* W_x     = (const float*)d_in[7];
    const float* W_dt    = (const float*)d_in[8];
    const float* b_dt    = (const float*)d_in[9];
    const float* Wc1     = (const float*)d_in[10];
    const float* bc1     = (const float*)d_in[11];
    const float* Wc2     = (const float*)d_in[12];
    const float* bc2     = (const float*)d_in[13];
    const float* A_log   = (const float*)d_in[14];
    const float* Dskip   = (const float*)d_in[15];
    const float* W_out   = (const float*)d_in[16];
    float* out = (float*)d_out;

    // bump allocator over workspace
    char*  ws  = (char*)d_ws;
    size_t off = 0;
    auto alloc = [&](size_t bytes) -> void* {
        void* p = ws + off;
        off = (off + bytes + 255) & ~(size_t)255;
        return p;
    };

    unsigned short* Winb  = (unsigned short*)alloc((size_t)D_MODEL * TWO_INNER * 2);
    unsigned short* Wc2b  = (unsigned short*)alloc((size_t)D_HID * D_INNER * 2);
    unsigned short* Woutb = (unsigned short*)alloc((size_t)D_INNER * D_MODEL * 2);
    unsigned short* xnb   = (unsigned short*)alloc((size_t)NROWS * D_MODEL * 2);
    float*          xzf   = (float*)alloc((size_t)NROWS * TWO_INNER * 4);
    float*          xcf   = (float*)alloc((size_t)NROWS * D_INNER * 4);
    float*          xssm  = (float*)alloc((size_t)NROWS * (2 * D_STATE + 1) * 4);
    unsigned short* Hb    = (unsigned short*)alloc((size_t)NROWS * D_HID * 2);
    float*          modb  = (float*)alloc((size_t)NROWS * D_INNER * 4);  // mod -> delta in place
    unsigned short* ygb   = (unsigned short*)alloc((size_t)NROWS * D_INNER * 2);

    // 1. weight conversions to bf16
    cvt_bf16_kernel<<<cdiv(D_MODEL * TWO_INNER, 256), 256, 0, stream>>>(W_in, Winb, D_MODEL * TWO_INNER);
    cvt_bf16_kernel<<<cdiv(D_HID * D_INNER, 256), 256, 0, stream>>>(Wc2, Wc2b, D_HID * D_INNER);
    cvt_bf16_kernel<<<cdiv(D_INNER * D_MODEL, 256), 256, 0, stream>>>(W_out, Woutb, D_INNER * D_MODEL);

    // 2. LayerNorm -> bf16
    layernorm_kernel<<<NROWS, 256, 0, stream>>>(x, ln_g, ln_b, xnb);

    // 3. xz = xn @ W_in   (4096 x 768 x 3072)
    wmma_gemm_kernel<TWO_INNER, D_MODEL, 0>
        <<<dim3(TWO_INNER / 128, NROWS / 128), 256, 0, stream>>>(xnb, Winb, xzf, nullptr);

    // 4. depthwise conv + SiLU -> xc
    conv_silu_kernel<<<cdiv(NROWS * D_INNER, 256), 256, 0, stream>>>(xzf, conv_w, conv_b, xcf);

    // 5. x_ssm = xc @ W_x (1536 -> 33)
    xssm_kernel<<<NROWS, 64, 0, stream>>>(xcf, W_x, xssm);

    // 6. H = relu(imp * Wc1 + bc1) -> bf16
    hproj_kernel<<<cdiv(NROWS * D_HID, 256), 256, 0, stream>>>(imp, Wc1, bc1, Hb);

    // 7. mod = sigmoid(H @ Wc2 + bc2)   (4096 x 384 x 1536)
    wmma_gemm_kernel<D_INNER, D_HID, 1>
        <<<dim3(D_INNER / 128, NROWS / 128), 256, 0, stream>>>(Hb, Wc2b, modb, bc2);

    // 8. delta = softplus(delta_raw*W_dt + b_dt) * (1 + mod), in place
    delta_kernel<<<cdiv(NROWS * D_INNER, 256), 256, 0, stream>>>(xssm, W_dt, b_dt, modb);

    // 9. selective scan -> yg (bf16), fused skip + SiLU(z) gating
    scan_kernel<<<(BATCH * D_INNER * D_STATE) / 256, 256, 0, stream>>>(
        modb, xssm, xcf, xzf, A_log, Dskip, ygb);

    // 10. out = yg @ W_out + residual   (4096 x 1536 x 768)
    wmma_gemm_kernel<D_MODEL, D_INNER, 2>
        <<<dim3(D_MODEL / 128, NROWS / 128), 256, 0, stream>>>(ygb, Woutb, out, x);
}
